// Discriminator_21680994910701
// MI455X (gfx1250) — matrix operand order repaired and checked by reading, not compiled, not run on hardware
//
#include <hip/hip_runtime.h>
#include <hip/hip_bf16.h>
#include <stdint.h>

typedef __attribute__((ext_vector_type(16))) _Float16 v16h;
typedef __attribute__((ext_vector_type(8)))  float    v8f;

union Frag16 { v16h v; uint32_t u[8]; };

// ---------------------------------------------------------------- utilities
__global__ void k_zero(float* __restrict__ p, long n) {
    long t = (long)blockIdx.x * blockDim.x + threadIdx.x;
    if (t < n) p[t] = 0.0f;
}

__global__ void k_deg(const int* __restrict__ dst, float* __restrict__ deg, int E) {
    int t = blockIdx.x * blockDim.x + threadIdx.x;
    if (t < E) atomicAdd(&deg[dst[t]], 1.0f);
}

__global__ void k_dinv(float* __restrict__ deg, int N) {
    int t = blockIdx.x * blockDim.x + threadIdx.x;
    if (t < N) {
        float d = deg[t];
        deg[t] = (d > 0.0f) ? rsqrtf(d) : 0.0f;
    }
}

__global__ void k_wn(const int* __restrict__ src, const int* __restrict__ dst,
                     const float* __restrict__ dinv, float* __restrict__ wn, int E) {
    int t = blockIdx.x * blockDim.x + threadIdx.x;
    if (t < E) wn[t] = dinv[src[t]] * dinv[dst[t]];
}

__global__ void k_cvt(const float* __restrict__ in, _Float16* __restrict__ out, long n) {
    long t = (long)blockIdx.x * blockDim.x + threadIdx.x;
    if (t < n) out[t] = (_Float16)in[t];
}

// W[k][d][n] (fp32) -> Wt[k][n][d] (f16), k<4, d,n<64
__global__ void k_convW(const float* __restrict__ W, _Float16* __restrict__ Wt) {
    int t = blockIdx.x * blockDim.x + threadIdx.x;
    if (t < 4 * 64 * 64) {
        int k = t >> 12;
        int d = (t >> 6) & 63;
        int n = t & 63;
        Wt[((k * 64 + n) << 6) + d] = (_Float16)W[t];
    }
}

// hout[dst] += wn[e] * hin[src]  over 64 features (hout pre-zeroed)
__global__ void k_spmm(const float* __restrict__ hin, const float* __restrict__ wn,
                       const int* __restrict__ src, const int* __restrict__ dst,
                       float* __restrict__ hout, int E) {
    long t = (long)blockIdx.x * blockDim.x + threadIdx.x;
    int e = (int)(t >> 4);
    if (e >= E) return;
    int f = (int)(t & 15) << 2;
    float w = wn[e];
    const float4 v = *reinterpret_cast<const float4*>(hin + (size_t)src[e] * 64 + f);
    float* o = hout + (size_t)dst[e] * 64 + f;
    atomicAdd(o + 0, w * v.x);
    atomicAdd(o + 1, w * v.y);
    atomicAdd(o + 2, w * v.z);
    atomicAdd(o + 3, w * v.w);
}

// ------------------------------------------------ fused TAGConv GEMM (WMMA)
// out[N][64] = PReLU( sum_{k<4} h16[k] @ Wt16[k]^T + bias , alpha )
// h16 : [4][N][64] f16 row-major,  Wt16 : [4][64(n)][64(d)] f16 (transposed)
__global__ __launch_bounds__(128)
void k_gemm_tag(const _Float16* __restrict__ h16,
                const _Float16* __restrict__ wt16,
                const float* __restrict__ bias,
                const float* __restrict__ alpha,
                float* __restrict__ out, int N) {
    __shared__ __align__(16) _Float16 ldsW[4 * 64 * 64]; // 32 KB
    __shared__ __align__(16) _Float16 ldsA[4 * 16 * 64]; //  8 KB

    const int tid  = threadIdx.x;
    const int lane = tid & 31;
    const int wv   = tid >> 5;              // wave = column tile 0..3

    // stage all 4 hop weight matrices once per block (2048 uint4)
    {
        const uint4* gw = reinterpret_cast<const uint4*>(wt16);
        uint4* lw = reinterpret_cast<uint4*>(ldsW);
        for (int i = tid; i < 2048; i += 128) lw[i] = gw[i];
    }

    const float a    = alpha[0];
    const float bcol = bias[(wv << 4) + (lane & 15)];
    const int   m    = lane & 15;
    const int   n    = (wv << 4) + m;
    const int   kbA  = (lane < 16) ? 0 : 8;
    const int   kbB  = (lane < 16) ? 0 : 8;
    const int   numRowTiles = N >> 4;       // N % 16 == 0 (N = 50000)

    for (int rt = blockIdx.x; rt < numRowTiles; rt += gridDim.x) {
        __syncthreads();                    // protects ldsA reuse + first W use
        // stage A: 4 hops x 16 rows x 64 halves (512 uint4)
        {
            const int rowBase = rt << 4;
            uint4* la = reinterpret_cast<uint4*>(ldsA);
            for (int i = tid; i < 512; i += 128) {
                int dg  = i & 7;
                int row = (i >> 3) & 15;
                int k   = i >> 7;
                la[i] = *reinterpret_cast<const uint4*>(
                    h16 + (((size_t)k * N + rowBase + row) << 6) + (dg << 3));
            }
        }
        __syncthreads();

        v8f c;
#pragma unroll
        for (int r = 0; r < 8; ++r) c[r] = bcol;

#pragma unroll
        for (int k = 0; k < 4; ++k) {
            const _Float16* Ak = ldsA + k * (16 * 64) + (m << 6);
            const _Float16* Bn = ldsW + k * (64 * 64) + (n << 6);
#pragma unroll
            for (int s = 0; s < 2; ++s) {   // K = 64 in two 32-deep steps
                Frag16 af, bf;
#pragma unroll
                for (int v = 0; v < 4; ++v) {
                    af.u[v]     = *reinterpret_cast<const uint32_t*>(Ak + s * 32 + kbA + 2 * v);
                    af.u[v + 4] = *reinterpret_cast<const uint32_t*>(Ak + s * 32 + 16 + kbA + 2 * v);
                }
#pragma unroll
                for (int v = 0; v < 4; ++v) {
                    bf.u[v]     = *reinterpret_cast<const uint32_t*>(Bn + s * 32 + kbB + 2 * v);
                    bf.u[v + 4] = *reinterpret_cast<const uint32_t*>(Bn + s * 32 + 16 + kbB + 2 * v);
                }

                c = __builtin_amdgcn_wmma_f32_16x16x32_f16(
                        false, af.v, false, bf.v, (short)0, c, false, false);
            }
        }

        // epilogue: PReLU + store (C layout: VGPR r -> row r (+8 for hi lanes))
        const int rowBase = (rt << 4) + ((lane < 16) ? 0 : 8);
#pragma unroll
        for (int r = 0; r < 8; ++r) {
            float vv = c[r];
            vv = (vv >= 0.0f) ? vv : a * vv;
            out[((size_t)(rowBase + r) << 6) + n] = vv;
        }
    }
}

// ---------------------------------------------------------------- pool+head
__global__ void k_pool(const float* __restrict__ h, const int* __restrict__ batch,
                       float* __restrict__ g, int N) {
    long t = (long)blockIdx.x * blockDim.x + threadIdx.x;
    if (t < (long)N * 64) {
        int node = (int)(t >> 6);
        int f    = (int)(t & 63);
        atomicAdd(&g[((size_t)batch[node] << 6) + f], h[t]);
    }
}

__global__ void k_head(const float* __restrict__ g, const float* __restrict__ Wout,
                       const float* __restrict__ bout, float* __restrict__ out, int G) {
    int t = blockIdx.x * blockDim.x + threadIdx.x;
    if (t < G) {
        float s = 0.0f;
        const float* gr = g + ((size_t)t << 6);
#pragma unroll
        for (int j = 0; j < 64; ++j) s += gr[j] * Wout[j];
        out[t] = s + bout[0];
    }
}

// ------------------------------------------------------------------- driver
extern "C" void kernel_launch(void* const* d_in, const int* in_sizes, int n_in,
                              void* d_out, int out_size, void* d_ws, size_t ws_size,
                              hipStream_t stream) {
    (void)n_in; (void)ws_size;
    const float* x    = (const float*)d_in[0];
    const int*   ei   = (const int*)d_in[1];
    const int*   bat  = (const int*)d_in[2];
    const float* W0   = (const float*)d_in[3];
    const float* b0   = (const float*)d_in[4];
    const float* W1   = (const float*)d_in[5];
    const float* b1   = (const float*)d_in[6];
    const float* a0   = (const float*)d_in[7];
    const float* a1   = (const float*)d_in[8];
    const float* Wo   = (const float*)d_in[9];
    const float* bo   = (const float*)d_in[10];
    float* out = (float*)d_out;

    const int N = in_sizes[2];
    const int E = in_sizes[1] / 2;
    const int G = out_size;            // 128 graphs
    const int* src = ei;
    const int* dst = ei + E;
    const long NF  = (long)N * 64;

    // carve workspace (256B aligned slices)
    char* ws = (char*)d_ws;
    size_t off = 0;
    auto take = [&](size_t bytes) -> char* {
        char* p = ws + off;
        off += (bytes + 255) & ~(size_t)255;
        return p;
    };
    float*    deg  = (float*)take((size_t)N * 4);
    float*    wn   = (float*)take((size_t)E * 4);
    float*    hA   = (float*)take((size_t)NF * 4);
    float*    hB   = (float*)take((size_t)NF * 4);
    float*    outL = (float*)take((size_t)NF * 4);
    _Float16* h16  = (_Float16*)take((size_t)4 * NF * 2);
    _Float16* W0t  = (_Float16*)take((size_t)4 * 64 * 64 * 2);
    _Float16* W1t  = (_Float16*)take((size_t)4 * 64 * 64 * 2);
    float*    gp   = (float*)take((size_t)G * 64 * 4);

    const int B = 256;
    auto cdiv = [](long a, long b) { return (int)((a + b - 1) / b); };

    // gcn_norm
    k_zero<<<cdiv(N, B), B, 0, stream>>>(deg, N);
    k_deg <<<cdiv(E, B), B, 0, stream>>>(dst, deg, E);
    k_dinv<<<cdiv(N, B), B, 0, stream>>>(deg, N);
    k_wn  <<<cdiv(E, B), B, 0, stream>>>(src, dst, deg, wn, E);

    // weight conversion (transposed, f16)
    k_convW<<<64, B, 0, stream>>>(W0, W0t);
    k_convW<<<64, B, 0, stream>>>(W1, W1t);

    // ---- layer 0 hops ----
    k_cvt <<<cdiv(NF, B), B, 0, stream>>>(x, h16, NF);
    k_zero<<<cdiv(NF, B), B, 0, stream>>>(hA, NF);
    k_spmm<<<cdiv((long)E * 16, B), B, 0, stream>>>(x, wn, src, dst, hA, E);
    k_cvt <<<cdiv(NF, B), B, 0, stream>>>(hA, h16 + NF, NF);
    k_zero<<<cdiv(NF, B), B, 0, stream>>>(hB, NF);
    k_spmm<<<cdiv((long)E * 16, B), B, 0, stream>>>(hA, wn, src, dst, hB, E);
    k_cvt <<<cdiv(NF, B), B, 0, stream>>>(hB, h16 + 2 * NF, NF);
    k_zero<<<cdiv(NF, B), B, 0, stream>>>(hA, NF);
    k_spmm<<<cdiv((long)E * 16, B), B, 0, stream>>>(hB, wn, src, dst, hA, E);
    k_cvt <<<cdiv(NF, B), B, 0, stream>>>(hA, h16 + 3 * NF, NF);
    k_gemm_tag<<<512, 128, 0, stream>>>(h16, W0t, b0, a0, outL, N);

    // ---- layer 1 hops ----
    k_cvt <<<cdiv(NF, B), B, 0, stream>>>(outL, h16, NF);
    k_zero<<<cdiv(NF, B), B, 0, stream>>>(hA, NF);
    k_spmm<<<cdiv((long)E * 16, B), B, 0, stream>>>(outL, wn, src, dst, hA, E);
    k_cvt <<<cdiv(NF, B), B, 0, stream>>>(hA, h16 + NF, NF);
    k_zero<<<cdiv(NF, B), B, 0, stream>>>(hB, NF);
    k_spmm<<<cdiv((long)E * 16, B), B, 0, stream>>>(hA, wn, src, dst, hB, E);
    k_cvt <<<cdiv(NF, B), B, 0, stream>>>(hB, h16 + 2 * NF, NF);
    k_zero<<<cdiv(NF, B), B, 0, stream>>>(hA, NF);
    k_spmm<<<cdiv((long)E * 16, B), B, 0, stream>>>(hB, wn, src, dst, hA, E);
    k_cvt <<<cdiv(NF, B), B, 0, stream>>>(hA, h16 + 3 * NF, NF);
    k_gemm_tag<<<512, 128, 0, stream>>>(h16, W1t, b1, a1, hB, N);   // layer-1 out -> hB

    // ---- pool + head ----
    k_zero<<<cdiv((long)G * 64, B), B, 0, stream>>>(gp, (long)G * 64);
    k_pool<<<cdiv(NF, B), B, 0, stream>>>(hB, bat, gp, N);
    k_head<<<1, G, 0, stream>>>(gp, Wo, bo, out, G);
}
